// LearnedRouter_14396730376577
// MI455X (gfx1250) — matrix operand order repaired and checked by reading, not compiled, not run on hardware
//
#include <hip/hip_runtime.h>
#include <hip/hip_bf16.h>

// ---------------------------------------------------------------------------
// MoE learned router for MI455X (gfx1250), compile-only tuned.
// Phase 1: logits = x @ W^T via bf16x3-split V_WMMA_F32_16X16X32_BF16.
//   Roofline: 2*16384*4096*64 = 8.6 GFLOP vs ~280 MB mandatory traffic
//   -> 12 us @ 23.3 TB/s. bf16 WMMA (16K FLOP/inst) keeps 3x-split compute
//   under that roof; f32 WMMA (2K FLOP/inst) would not.
//   Split: hi = trunc16(f) (1 v_perm), residual exact (Sterbenz), lo =
//   round-half-up bf16 of residual (add + v_perm) -> ~7 VALU per 2 floats.
// Phase 2: softmax + expert_mask + top-2 + softmax + L1-renorm (trivial).
// ---------------------------------------------------------------------------

typedef __attribute__((ext_vector_type(16))) __bf16    v16bf;
typedef __attribute__((ext_vector_type(8)))  float     v8f;
typedef __attribute__((ext_vector_type(8)))  unsigned  v8u;

#define HDIM   4096
#define NEXP   64
#define KSUP   128          // K-superchunk staged in LDS
#define PITCH  68           // dwords per expert row in LDS (64 data + 4 pad -> bank stride 4)

// split (a,b) into packed bf16 hi (truncated) and packed bf16 lo (residual).
// hi truncation makes the residual exact; lo uses round-half-up (unbiased
// except on exact ties). Effective combined mantissa ~16 bits.
__device__ __forceinline__ void split2(float a, float b, unsigned* h, unsigned* l) {
  unsigned ua = __float_as_uint(a), ub = __float_as_uint(b);
  *h = __builtin_amdgcn_perm(ub, ua, 0x07060302u);        // {b[31:16], a[31:16]}
  float ra = a - __uint_as_float(ua & 0xffff0000u);       // exact residuals
  float rb = b - __uint_as_float(ub & 0xffff0000u);
  unsigned la = __float_as_uint(ra) + 0x8000u;            // round-half-up
  unsigned lb = __float_as_uint(rb) + 0x8000u;
  *l = __builtin_amdgcn_perm(lb, la, 0x07060302u);
}

__global__ __launch_bounds__(256) void router_gemm_bf16x3(
    const float* __restrict__ x,      // [rows, HDIM]
    const float* __restrict__ W,      // [NEXP, HDIM]
    float* __restrict__ logits)       // [rows, NEXP]
{
  __shared__ unsigned lds_hi[NEXP * PITCH];
  __shared__ unsigned lds_lo[NEXP * PITCH];

  const int tid    = threadIdx.x;
  const int lane   = tid & 31;
  const int wave   = tid >> 5;
  const int row0   = (blockIdx.x * 8 + wave) * 16;   // 16-row strip per wave
  const int lane15 = lane & 15;
  const int mhalf  = (lane >> 4) << 3;               // +8 rows for upper half (C layout)
  const int aoff   = (lane >> 4) << 3;               // A: upper lanes take K+8
  const int boff   = (lane >> 4) << 3;               // B: upper lanes take K+8

  // staging map: 4 threads per expert, 32 K each
  const int se = tid >> 2;
  const int sk = (tid & 3) << 5;

  v8f acc[4];
#pragma unroll
  for (int t = 0; t < 4; ++t) acc[t] = (v8f)0.0f;

  const float* xrow = x + (size_t)(row0 + lane15) * HDIM;

  for (int ks = 0; ks < HDIM; ks += KSUP) {
    __syncthreads();
    // ---- stage W[:, ks..ks+KSUP) into LDS as split bf16 hi/lo packs ----
    {
      const float* wp = W + (size_t)se * HDIM + ks + sk;
      unsigned* ph = &lds_hi[se * PITCH + (sk >> 1)];
      unsigned* pl = &lds_lo[se * PITCH + (sk >> 1)];
#pragma unroll
      for (int j = 0; j < 8; ++j) {
        float4 f = *(const float4*)(wp + 4 * j);
        unsigned h0, l0, h1, l1;
        split2(f.x, f.y, &h0, &l0);
        split2(f.z, f.w, &h1, &l1);
        ph[2 * j]     = h0;  ph[2 * j + 1] = h1;
        pl[2 * j]     = l0;  pl[2 * j + 1] = l1;
      }
    }
    __syncthreads();

#pragma unroll
    for (int kc = 0; kc < KSUP; kc += 32) {
      // ---- A operand: 16x32 bf16 layout (lanes 0-15: K[0:8)+K[16:24),
      //      lanes 16-31: K[8:16)+K[24:32)) -> wave covers full 128B lines ----
      const float* xp = xrow + ks + kc + aoff;
      float4 f0 = *(const float4*)(xp + 0);
      float4 f1 = *(const float4*)(xp + 4);
      float4 f2 = *(const float4*)(xp + 16);
      float4 f3 = *(const float4*)(xp + 20);
      unsigned ahv[8], alv[8];
      split2(f0.x, f0.y, &ahv[0], &alv[0]);
      split2(f0.z, f0.w, &ahv[1], &alv[1]);
      split2(f1.x, f1.y, &ahv[2], &alv[2]);
      split2(f1.z, f1.w, &ahv[3], &alv[3]);
      split2(f2.x, f2.y, &ahv[4], &alv[4]);
      split2(f2.z, f2.w, &ahv[5], &alv[5]);
      split2(f3.x, f3.y, &ahv[6], &alv[6]);
      split2(f3.z, f3.w, &ahv[7], &alv[7]);
      v8u ahp = {ahv[0], ahv[1], ahv[2], ahv[3], ahv[4], ahv[5], ahv[6], ahv[7]};
      v8u alp = {alv[0], alv[1], alv[2], alv[3], alv[4], alv[5], alv[6], alv[7]};
      v16bf Ahi = __builtin_bit_cast(v16bf, ahp);
      v16bf Alo = __builtin_bit_cast(v16bf, alp);

#pragma unroll
      for (int t = 0; t < 4; ++t) {
        // B operand: 32x16 bf16, lane = expert column, lanes 16-31 take K+8 (and K+24).
        // pitch 68 dwords -> lane-to-lane bank stride 4: conflict-free b128.
        int uoff = ((t << 4) + lane15) * PITCH + ((kc + boff) >> 1);
        const uint4* bh = (const uint4*)&lds_hi[uoff];
        const uint4* bl = (const uint4*)&lds_lo[uoff];
        uint4 h0 = bh[0], h1 = bh[2];
        uint4 l0 = bl[0], l1 = bl[2];
        v8u hp = {h0.x, h0.y, h0.z, h0.w, h1.x, h1.y, h1.z, h1.w};
        v8u lp = {l0.x, l0.y, l0.z, l0.w, l1.x, l1.y, l1.z, l1.w};
        v16bf Bhi = __builtin_bit_cast(v16bf, hp);
        v16bf Blo = __builtin_bit_cast(v16bf, lp);
        // bf16x3 split product: hi*hi + hi*lo + lo*hi (~fp32 accuracy)
        acc[t] = __builtin_amdgcn_wmma_f32_16x16x32_bf16(false, Ahi, false, Bhi,
                                                         (short)0, acc[t], false, false);
        acc[t] = __builtin_amdgcn_wmma_f32_16x16x32_bf16(false, Ahi, false, Blo,
                                                         (short)0, acc[t], false, false);
        acc[t] = __builtin_amdgcn_wmma_f32_16x16x32_bf16(false, Alo, false, Bhi,
                                                         (short)0, acc[t], false, false);
      }
    }
  }

  // ---- epilogue: C layout -> lane = N, VGPR r = row (+8 for lanes 16-31) ----
#pragma unroll
  for (int t = 0; t < 4; ++t) {
    int n = (t << 4) + lane15;
#pragma unroll
    for (int r = 0; r < 8; ++r) {
      int row = row0 + r + mhalf;
      logits[(size_t)row * NEXP + n] = acc[t][r];
    }
  }
}

__global__ __launch_bounds__(256) void router_softmax_topk(
    const float* __restrict__ logits,       // [rows, NEXP]
    const float* __restrict__ expert_mask,  // [bs, NEXP]
    float* __restrict__ scores,             // [rows, NEXP]
    float* __restrict__ wts,                // [rows, 2]
    float* __restrict__ idxs,               // [rows, 2] (indices as float)
    int rows)
{
  int row = blockIdx.x * blockDim.x + threadIdx.x;
  if (row >= rows) return;
  const int b = row >> 12;   // sq = 4096 rows per batch element

  float s[NEXP];
  const float4* lp = (const float4*)(logits + (size_t)row * NEXP);
#pragma unroll
  for (int j = 0; j < NEXP / 4; ++j) {
    float4 v = lp[j];
    s[4 * j] = v.x; s[4 * j + 1] = v.y; s[4 * j + 2] = v.z; s[4 * j + 3] = v.w;
  }

  float mx = s[0];
#pragma unroll
  for (int e = 1; e < NEXP; ++e) mx = fmaxf(mx, s[e]);
  float sum = 0.0f;
#pragma unroll
  for (int e = 0; e < NEXP; ++e) { s[e] = __expf(s[e] - mx); sum += s[e]; }
  float inv = 1.0f / sum;

  const float* mrow = expert_mask + (size_t)b * NEXP;
#pragma unroll
  for (int e = 0; e < NEXP; ++e) s[e] = mrow[e] * s[e] * inv;

  float4* sp = (float4*)(scores + (size_t)row * NEXP);
#pragma unroll
  for (int j = 0; j < NEXP / 4; ++j) {
    float4 v; v.x = s[4 * j]; v.y = s[4 * j + 1]; v.z = s[4 * j + 2]; v.w = s[4 * j + 3];
    sp[j] = v;
  }

  // top-2 (first occurrence wins ties, matching lax.top_k)
  float s0 = -3.0e38f; int i0 = 0;
#pragma unroll
  for (int e = 0; e < NEXP; ++e) if (s[e] > s0) { s0 = s[e]; i0 = e; }
  float s1 = -3.0e38f; int i1 = 0;
#pragma unroll
  for (int e = 0; e < NEXP; ++e) if (e != i0 && s[e] > s1) { s1 = s[e]; i1 = e; }

  // softmax over the two picked score values, then L1 renorm (identity here)
  float e1 = __expf(s1 - s0);
  float den = 1.0f + e1;
  float w0 = 1.0f / den, w1 = e1 / den;
  float l1n = 1.0f / (fabsf(w0) + fabsf(w1));
  w0 *= l1n; w1 *= l1n;

  wts[(size_t)row * 2 + 0]  = w0;
  wts[(size_t)row * 2 + 1]  = w1;
  idxs[(size_t)row * 2 + 0] = (float)i0;
  idxs[(size_t)row * 2 + 1] = (float)i1;
}

extern "C" void kernel_launch(void* const* d_in, const int* in_sizes, int n_in,
                              void* d_out, int out_size, void* d_ws, size_t ws_size,
                              hipStream_t stream) {
  const float* x    = (const float*)d_in[0];   // [4,4096,4096] fp32
  const float* mask = (const float*)d_in[1];   // [4,64] fp32
  const float* W    = (const float*)d_in[2];   // [64,4096] fp32

  const int rows = in_sizes[0] / HDIM;         // 16384

  float* out     = (float*)d_out;
  float* scores  = out;                                   // rows*NEXP
  float* logits  = scores + (size_t)rows * NEXP;          // rows*NEXP
  float* weights = logits + (size_t)rows * NEXP;          // rows*2
  float* indices = weights + (size_t)rows * 2;            // rows*2 (as float)

  // 8 waves/block, 16 rows per wave -> 128 rows per block
  router_gemm_bf16x3<<<rows / 128, 256, 0, stream>>>(x, W, logits);
  router_softmax_topk<<<(rows + 255) / 256, 256, 0, stream>>>(
      logits, mask, scores, weights, indices, rows);
}
